// ProtoDINO_36515811951237
// MI455X (gfx1250) — matrix operand order repaired and checked by reading, not compiled, not run on hardware
//
#include <hip/hip_runtime.h>
#include <hip/hip_bf16.h>
#include <math.h>
#include <float.h>

// ---------------- problem constants ----------------
#define B_IMG   64
#define NPATCH  256
#define NTOK    (B_IMG * NPATCH)     // 16384
#define DIM     768
#define C_CLS   201
#define KPROT   5
#define CK      (C_CLS * KPROT)      // 1005
#define CKP     1024                 // padded to 64 x 16-col tiles (for 64-wide N blocking)
#define TEMP_INV 5.0f                // 1/0.2
#define GAMMA_   0.999f
#define SK_INV_EPS 20.0f             // 1/0.05

typedef __attribute__((ext_vector_type(16))) __bf16 v16bf;
typedef __attribute__((ext_vector_type(8)))  __bf16 v8bf;
typedef __attribute__((ext_vector_type(8)))  float  v8f;

// ---------------- helpers ----------------
__device__ __forceinline__ float atomicMaxF(float* addr, float val) {
    int* ai = (int*)addr;
    int old = *ai;
    while (__int_as_float(old) < val) {
        int assumed = old;
        old = atomicCAS(ai, assumed, __float_as_int(val));
        if (old == assumed) break;
    }
    return __int_as_float(old);
}

// ---------------- prep: token norms + bf16 convert ----------------
__global__ __launch_bounds__(256) void token_prep(const float* __restrict__ tok,
                                                  __bf16* __restrict__ tokbf,
                                                  float* __restrict__ inv_norm) {
    int n = blockIdx.x, t = threadIdx.x;
    const float* rowp = tok + (size_t)n * DIM;
    float v[3]; float ss = 0.f;
    #pragma unroll
    for (int i = 0; i < 3; ++i) { float x = rowp[t + 256 * i]; v[i] = x; ss += x * x; }
    #pragma unroll
    for (int off = 16; off > 0; off >>= 1) ss += __shfl_down(ss, off, 32);
    __shared__ float red[8];
    __shared__ float stot;
    if ((t & 31) == 0) red[t >> 5] = ss;
    __syncthreads();
    if (t == 0) {
        float s = 0.f;
        #pragma unroll
        for (int i = 0; i < 8; ++i) s += red[i];
        stot = 1.0f / fmaxf(sqrtf(s), 1e-12f);
    }
    __syncthreads();
    #pragma unroll
    for (int i = 0; i < 3; ++i) tokbf[(size_t)n * DIM + t + 256 * i] = (__bf16)v[i];
    if (t == 0) inv_norm[n] = stot;
}

// one block per padded (c,k) row; rows >= CK are zero-filled
__global__ __launch_bounds__(256) void proto_prep(const float* __restrict__ protos,
                                                  __bf16* __restrict__ pbf) {
    int r = blockIdx.x, t = threadIdx.x;
    if (r >= CK) {
        #pragma unroll
        for (int i = 0; i < 3; ++i) pbf[(size_t)r * DIM + t + 256 * i] = (__bf16)0.0f;
        return;
    }
    const float* rowp = protos + (size_t)r * DIM;
    float v[3]; float ss = 0.f;
    #pragma unroll
    for (int i = 0; i < 3; ++i) { float x = rowp[t + 256 * i]; v[i] = x; ss += x * x; }
    #pragma unroll
    for (int off = 16; off > 0; off >>= 1) ss += __shfl_down(ss, off, 32);
    __shared__ float red[8];
    __shared__ float stot;
    if ((t & 31) == 0) red[t >> 5] = ss;
    __syncthreads();
    if (t == 0) {
        float s = 0.f;
        #pragma unroll
        for (int i = 0; i < 8; ++i) s += red[i];
        stot = 1.0f / fmaxf(sqrtf(s), 1e-12f);
    }
    __syncthreads();
    #pragma unroll
    for (int i = 0; i < 3; ++i)
        pbf[(size_t)r * DIM + t + 256 * i] = (__bf16)(v[i] * stot);
}

// ---------------- main WMMA GEMM ----------------
// logits[n, ck] = tokens[n,:] . protos_n[ck,:]
// wave computes a 32x64 macro-tile (2 M-tiles x 4 N-tiles):
//   per k-step: 2 A fragments + 4 B fragments (12 x b128) feed 8 v_wmma_f32_16x16x32_bf16
__global__ __launch_bounds__(256) void gemm_logits(const __bf16* __restrict__ Abf,
                                                   const __bf16* __restrict__ Bbf,
                                                   float* __restrict__ out) {
    int lane = threadIdx.x & 31;
    int wid  = blockIdx.x * 8 + (threadIdx.x >> 5);  // global wave id
    int ng   = wid & 15;        // 0..15  (64-col groups over CKP=1024)
    int mg   = wid >> 4;        // 0..511 (32-row groups over NTOK=16384)
    int half = lane >> 4;       // 0/1
    int l15  = lane & 15;
    int sel8  = half ? 8 : 0;
    int sel16 = half ? 16 : 0;

    const __bf16* arow0 = Abf + (size_t)(mg * 32 + l15) * DIM;        // A tile i=0: M = l15
    const __bf16* arow1 = arow0 + (size_t)16 * DIM;                   // A tile i=1: M = 16 + l15
    const __bf16* brow[4];
    #pragma unroll
    for (int j = 0; j < 4; ++j)
        brow[j] = Bbf + (size_t)(ng * 64 + j * 16 + l15) * DIM;       // B col = l15 within tile j

    v8f acc[2][4];
    #pragma unroll
    for (int i = 0; i < 2; ++i)
        #pragma unroll
        for (int j = 0; j < 4; ++j) acc[i][j] = (v8f){};

    #pragma unroll 2
    for (int kb = 0; kb < DIM; kb += 32) {
        union { v16bf v; v8bf h[2]; } a[2], b[4];
        // A 16x32 bf16 fragment: lanes<16 K={0..7,16..23}, lanes>=16 K={8..15,24..31}
        a[0].h[0] = *(const v8bf*)(arow0 + kb + sel8);
        a[0].h[1] = *(const v8bf*)(arow0 + kb + 16 + sel8);
        a[1].h[0] = *(const v8bf*)(arow1 + kb + sel8);
        a[1].h[1] = *(const v8bf*)(arow1 + kb + 16 + sel8);
        // B 32x16 fragment: lanes<16 K={0..7,16..23}, lanes>=16 K={8..15,24..31}
        #pragma unroll
        for (int j = 0; j < 4; ++j) {
            b[j].h[0] = *(const v8bf*)(brow[j] + kb + sel8);
            b[j].h[1] = *(const v8bf*)(brow[j] + kb + 16 + sel8);
        }
        #pragma unroll
        for (int i = 0; i < 2; ++i)
            #pragma unroll
            for (int j = 0; j < 4; ++j)
                acc[i][j] = __builtin_amdgcn_wmma_f32_16x16x32_bf16(
                    false, a[i].v, false, b[j].v, (short)0, acc[i][j], false, false);
    }

    // epilogue: C/D layout -> VGPR r: lanes<16 M=r, lanes>=16 M=8+r; N = l15
    #pragma unroll
    for (int j = 0; j < 4; ++j) {
        int outcol = ng * 64 + j * 16 + l15;
        if (outcol < CK) {
            #pragma unroll
            for (int i = 0; i < 2; ++i) {
                int mbase = mg * 32 + i * 16 + half * 8;
                #pragma unroll
                for (int r = 0; r < 8; ++r)
                    out[(size_t)(mbase + r) * CK + outcol] = acc[i][j][r];
            }
        }
    }
}

// ---------------- per-image max over patches (logits and sims = logits*inv_norm) --------
__global__ __launch_bounds__(256) void image_max(const float* __restrict__ logits,
                                                 const float* __restrict__ inv_norm,
                                                 float* __restrict__ img_logits,
                                                 float* __restrict__ img_sims) {
    int b = blockIdx.y;
    int ck = blockIdx.x * 256 + threadIdx.x;
    if (ck >= CK) return;
    const float* base = logits + (size_t)b * NPATCH * CK + ck;
    const float* invp = inv_norm + b * NPATCH;
    float ml = -FLT_MAX, ms = -FLT_MAX;
    for (int p = 0; p < NPATCH; ++p) {
        float v = base[(size_t)p * CK];
        ml = fmaxf(ml, v);
        ms = fmaxf(ms, v * invp[p]);
    }
    img_logits[b * CK + ck] = ml;
    img_sims[b * CK + ck]   = ms;
}

// ---------------- ScoreAggregation head ----------------
__global__ __launch_bounds__(256) void class_head(const float* __restrict__ img_logits,
                                                  const float* __restrict__ sa_w,
                                                  float* __restrict__ out) {
    int id = blockIdx.x * 256 + threadIdx.x;
    if (id >= B_IMG * (C_CLS - 1)) return;
    int b = id / (C_CLS - 1), c = id % (C_CLS - 1);
    float w[KPROT]; float mx = -FLT_MAX;
    #pragma unroll
    for (int k = 0; k < KPROT; ++k) { w[k] = sa_w[c * KPROT + k]; mx = fmaxf(mx, w[k]); }
    float se = 0.f;
    #pragma unroll
    for (int k = 0; k < KPROT; ++k) { w[k] = __expf(w[k] - mx); se += w[k]; }
    float acc = 0.f;
    #pragma unroll
    for (int k = 0; k < KPROT; ++k)
        acc += img_logits[(size_t)b * CK + c * KPROT + k] * (w[k] / se * (float)KPROT);
    out[id] = acc * TEMP_INV;
}

// ---------------- small-buffer init ----------------
__global__ void init_small(float* s_fg, float* s_bg, float* m_c, float* S_c,
                           int* n_c, float* rbuf) {
    int t = blockIdx.x * 256 + threadIdx.x;
    if (t < B_IMG) { s_fg[t] = 0.f; s_bg[t] = 0.f; }
    if (t < C_CLS) { m_c[t] = -FLT_MAX; S_c[t] = 0.f; n_c[t] = 0; }
    if (t < CK)    { rbuf[t] = 0.f; }
}

__global__ void zero_f32(float* p, int n) {
    int i = blockIdx.x * 256 + threadIdx.x;
    if (i < n) p[i] = 0.f;
}

// ---------------- pseudo labels ----------------
__global__ __launch_bounds__(256) void fg_pass1(const float* __restrict__ logits,
                                                const int* __restrict__ labels,
                                                float* __restrict__ fgv, float* __restrict__ bgv,
                                                float* __restrict__ s_fg, float* __restrict__ s_bg) {
    int n = blockIdx.x * 256 + threadIdx.x;
    if (n >= NTOK) return;
    int b = n / NPATCH;
    int lab = labels[b];
    const float* row = logits + (size_t)n * CK;
    float f = 0.f, g = 0.f;
    #pragma unroll
    for (int k = 0; k < KPROT; ++k) { f += row[lab * KPROT + k]; g += row[(C_CLS - 1) * KPROT + k]; }
    fgv[n] = f; bgv[n] = g;
    atomicAdd(&s_fg[b], fabsf(f));
    atomicAdd(&s_bg[b], fabsf(g));
}

__global__ __launch_bounds__(256) void fg_pass2(const int* __restrict__ labels,
                                                const float* __restrict__ fgv, const float* __restrict__ bgv,
                                                const float* __restrict__ s_fg, const float* __restrict__ s_bg,
                                                int* __restrict__ pseudo, float* __restrict__ pseudo_out) {
    int n = blockIdx.x * 256 + threadIdx.x;
    if (n >= NTOK) return;
    int b = n / NPATCH;
    float f = fgv[n] / fmaxf(s_fg[b], 1e-12f);
    float g = bgv[n] / fmaxf(s_bg[b], 1e-12f);
    int p = (f > g) ? labels[b] : (C_CLS - 1);   // argmax([bg,fg]): bg wins ties
    pseudo[n] = p;
    pseudo_out[n] = (float)p;
}

// ---------------- Sinkhorn (compact [N,K] representation) ----------------
__global__ __launch_bounds__(256) void sk_gather(const float* __restrict__ logits,
                                                 const int* __restrict__ pseudo,
                                                 float* __restrict__ Ac,
                                                 float* __restrict__ m_c, int* __restrict__ n_c) {
    int n = blockIdx.x * 256 + threadIdx.x;
    if (n >= NTOK) return;
    int c = pseudo[n];
    const float* row = logits + (size_t)n * CK + c * KPROT;
    float mx = -FLT_MAX;
    #pragma unroll
    for (int k = 0; k < KPROT; ++k) { float v = row[k]; Ac[n * KPROT + k] = v; mx = fmaxf(mx, v); }
    atomicMaxF(&m_c[c], mx);
    atomicAdd(&n_c[c], 1);
}

__global__ void sk_fix_m(float* m_c) {
    int c = blockIdx.x * 256 + threadIdx.x;
    if (c < C_CLS && m_c[c] == -FLT_MAX) m_c[c] = 0.f;   // no members: m -> 0
}

__global__ __launch_bounds__(256) void sk_exp(float* __restrict__ Ac,
                                              const int* __restrict__ pseudo,
                                              const float* __restrict__ m_c,
                                              float* __restrict__ S_c) {
    int n = blockIdx.x * 256 + threadIdx.x;
    if (n >= NTOK) return;
    int c = pseudo[n];
    float m = m_c[c], s = 0.f;
    #pragma unroll
    for (int k = 0; k < KPROT; ++k) {
        float e = __expf((Ac[n * KPROT + k] - m) * SK_INV_EPS);
        Ac[n * KPROT + k] = e; s += e;
    }
    atomicAdd(&S_c[c], s);
}

__global__ __launch_bounds__(256) void sk_gnorm(float* __restrict__ Ac,
                                                const int* __restrict__ pseudo,
                                                const float* __restrict__ S_c) {
    int n = blockIdx.x * 256 + threadIdx.x;
    if (n >= NTOK) return;
    float S = S_c[pseudo[n]];
    float inv = 1.0f / (S > 0.f ? S : 1.0f);
    #pragma unroll
    for (int k = 0; k < KPROT; ++k) Ac[n * KPROT + k] *= inv;
}

__global__ __launch_bounds__(256) void sk_sumr(const float* __restrict__ Ac,
                                               const int* __restrict__ pseudo,
                                               float* __restrict__ rbuf) {
    int n = blockIdx.x * 256 + threadIdx.x;
    if (n >= NTOK) return;
    int c = pseudo[n];
    #pragma unroll
    for (int k = 0; k < KPROT; ++k) atomicAdd(&rbuf[c * KPROT + k], Ac[n * KPROT + k]);
}

__global__ __launch_bounds__(256) void sk_apply(float* __restrict__ Ac,
                                                const int* __restrict__ pseudo,
                                                const float* __restrict__ rbuf,
                                                const int* __restrict__ n_c) {
    int n = blockIdx.x * 256 + threadIdx.x;
    if (n >= NTOK) return;
    int c = pseudo[n];
    float ncf = n_c[c] > 0 ? (float)n_c[c] : 1.0f;
    float a[KPROT]; float s = 0.f;
    #pragma unroll
    for (int k = 0; k < KPROT; ++k) {
        float r = rbuf[c * KPROT + k];
        float v = Ac[n * KPROT + k] / ((r > 0.f ? r : 1.0f) * (float)KPROT);
        a[k] = v; s += v;
    }
    float invs = 1.0f / ((s > 0.f ? s : 1.0f) * ncf);
    #pragma unroll
    for (int k = 0; k < KPROT; ++k) Ac[n * KPROT + k] = a[k] * invs;
}

__global__ __launch_bounds__(256) void sk_assign(const float* __restrict__ Ac,
                                                 float* __restrict__ part_out) {
    int n = blockIdx.x * 256 + threadIdx.x;
    if (n >= NTOK) return;
    float best = Ac[n * KPROT]; int bi = 0;
    #pragma unroll
    for (int k = 1; k < KPROT; ++k) {
        float v = Ac[n * KPROT + k];
        if (v > best) { best = v; bi = k; }
    }
    part_out[n] = (float)bi;
}

// masked scatter-GEMM: P_raw[c,k,d] += (A*n_c)[n,k] * tokens_n[n,d]; block per patch
__global__ __launch_bounds__(256) void sk_scatter(const float* __restrict__ Ac,
                                                  const int* __restrict__ pseudo,
                                                  const int* __restrict__ n_c,
                                                  const float* __restrict__ tok,
                                                  const float* __restrict__ inv_norm,
                                                  float* __restrict__ P_raw) {
    int n = blockIdx.x, t = threadIdx.x;
    int c = pseudo[n];
    float ncf = (float)n_c[c];
    float a[KPROT];
    #pragma unroll
    for (int k = 0; k < KPROT; ++k) a[k] = Ac[n * KPROT + k] * ncf;
    float iv = inv_norm[n];
    #pragma unroll
    for (int i = 0; i < 3; ++i) {
        int d = t + 256 * i;
        float x = tok[(size_t)n * DIM + d] * iv;
        #pragma unroll
        for (int k = 0; k < KPROT; ++k)
            atomicAdd(&P_raw[((size_t)c * KPROT + k) * DIM + d], a[k] * x);
    }
}

__global__ __launch_bounds__(256) void new_protos(const float* __restrict__ protos,
                                                  const float* __restrict__ P_raw,
                                                  const int* __restrict__ n_c,
                                                  float* __restrict__ out) {
    int i = blockIdx.x * 256 + threadIdx.x;
    if (i >= CK * DIM) return;
    int c = i / (KPROT * DIM);
    float p = protos[i];
    out[i] = (n_c[c] > 0) ? (GAMMA_ * p + (1.0f - GAMMA_) * P_raw[i]) : p;
}

// ---------------- host launch ----------------
extern "C" void kernel_launch(void* const* d_in, const int* in_sizes, int n_in,
                              void* d_out, int out_size, void* d_ws, size_t ws_size,
                              hipStream_t stream) {
    const float* patch_tokens = (const float*)d_in[0];   // [64,256,768]
    const float* prototypes   = (const float*)d_in[1];   // [201,5,768]
    const float* sa_weights   = (const float*)d_in[2];   // [200,5]
    const int*   labels       = (const int*)d_in[3];     // [64]

    // output regions (concatenated flat, f32)
    float* out = (float*)d_out;
    float* o_class_logits = out;                                   // 64*200
    float* o_img_logits   = o_class_logits + B_IMG * (C_CLS - 1);  // 64*1005
    float* o_patch_logits = o_img_logits + B_IMG * CK;             // 16384*1005
    float* o_img_sims     = o_patch_logits + (size_t)NTOK * CK;    // 64*1005
    float* o_part_assign  = o_img_sims + B_IMG * CK;               // 16384
    float* o_pseudo       = o_part_assign + NTOK;                  // 16384
    float* o_new_protos   = o_pseudo + NTOK;                       // 201*5*768

    // workspace carve
    char* w = (char*)d_ws;
    auto carve = [&](size_t bytes) { char* p = w; w += (bytes + 255) & ~(size_t)255; return (void*)p; };
    __bf16* tokbf    = (__bf16*)carve((size_t)NTOK * DIM * sizeof(__bf16));
    __bf16* pbf      = (__bf16*)carve((size_t)CKP * DIM * sizeof(__bf16));
    float*  inv_norm = (float*)carve((size_t)NTOK * sizeof(float));
    float*  Ac       = (float*)carve((size_t)NTOK * KPROT * sizeof(float));
    float*  P_raw    = (float*)carve((size_t)CK * DIM * sizeof(float));
    float*  fgv      = (float*)carve((size_t)NTOK * sizeof(float));
    float*  bgv      = (float*)carve((size_t)NTOK * sizeof(float));
    int*    pseudo_i = (int*)carve((size_t)NTOK * sizeof(int));
    float*  s_fg     = (float*)carve(B_IMG * sizeof(float));
    float*  s_bg     = (float*)carve(B_IMG * sizeof(float));
    float*  m_c      = (float*)carve(C_CLS * sizeof(float));
    float*  S_c      = (float*)carve(C_CLS * sizeof(float));
    int*    n_c      = (int*)carve(C_CLS * sizeof(int));
    float*  rbuf     = (float*)carve(CK * sizeof(float));

    // 1) prep
    token_prep<<<NTOK, 256, 0, stream>>>(patch_tokens, tokbf, inv_norm);
    proto_prep<<<CKP, 256, 0, stream>>>(prototypes, pbf);

    // 2) one fused bf16 WMMA GEMM (logits; sims derived by row scaling)
    //    waves: (16384/32) * (1024/64) = 512 * 16 = 8192 -> 1024 blocks of 8 waves
    gemm_logits<<<(NTOK / 32) * (CKP / 64) / 8, 256, 0, stream>>>(tokbf, pbf, o_patch_logits);

    // 3) per-image maxima + classification head
    image_max<<<dim3((CK + 255) / 256, B_IMG), 256, 0, stream>>>(o_patch_logits, inv_norm,
                                                                 o_img_logits, o_img_sims);
    class_head<<<(B_IMG * (C_CLS - 1) + 255) / 256, 256, 0, stream>>>(o_img_logits, sa_weights,
                                                                      o_class_logits);

    // 4) small-state init
    init_small<<<8, 256, 0, stream>>>(s_fg, s_bg, m_c, S_c, n_c, rbuf);
    zero_f32<<<(CK * DIM + 255) / 256, 256, 0, stream>>>(P_raw, CK * DIM);

    // 5) pseudo labels
    fg_pass1<<<NTOK / 256, 256, 0, stream>>>(o_patch_logits, labels, fgv, bgv, s_fg, s_bg);
    fg_pass2<<<NTOK / 256, 256, 0, stream>>>(labels, fgv, bgv, s_fg, s_bg, pseudo_i, o_pseudo);

    // 6) per-class Sinkhorn on compact [N,K]
    sk_gather<<<NTOK / 256, 256, 0, stream>>>(o_patch_logits, pseudo_i, Ac, m_c, n_c);
    sk_fix_m<<<1, 256, 0, stream>>>(m_c);
    sk_exp<<<NTOK / 256, 256, 0, stream>>>(Ac, pseudo_i, m_c, S_c);
    sk_gnorm<<<NTOK / 256, 256, 0, stream>>>(Ac, pseudo_i, S_c);
    for (int it = 0; it < 3; ++it) {
        zero_f32<<<(CK + 255) / 256, 256, 0, stream>>>(rbuf, CK);
        sk_sumr<<<NTOK / 256, 256, 0, stream>>>(Ac, pseudo_i, rbuf);
        sk_apply<<<NTOK / 256, 256, 0, stream>>>(Ac, pseudo_i, rbuf, n_c);
    }
    sk_assign<<<NTOK / 256, 256, 0, stream>>>(Ac, o_part_assign);

    // 7) prototype update (masked scatter-GEMM + EMA)
    sk_scatter<<<NTOK, 256, 0, stream>>>(Ac, pseudo_i, n_c, patch_tokens, inv_norm, P_raw);
    new_protos<<<(CK * DIM + 255) / 256, 256, 0, stream>>>(prototypes, P_raw, n_c, o_new_protos);

    (void)in_sizes; (void)n_in; (void)out_size; (void)ws_size;
}
